// SimpleLSTM_64957085385390
// MI455X (gfx1250) — matrix operand order repaired and checked by reading, not compiled, not run on hardware
//
#include <hip/hip_runtime.h>

typedef __attribute__((ext_vector_type(16))) _Float16 v16h;
typedef __attribute__((ext_vector_type(8)))  _Float16 v8h;
typedef __attribute__((ext_vector_type(8)))  float    v8f;

#define BATCH 1024
#define TSTEPS 512
#define HID 32
#define NCLS 2
#define HPITCH 40   // f16 per LDS row (padded from 32 -> conflict-free banks)

// gfx1250 hardware v_tanh_f32 transcendental (confirmed in round-2 disasm).
#if __has_builtin(__builtin_amdgcn_tanhf)
__device__ __forceinline__ float tanh_hw(float x) {
    return __builtin_amdgcn_tanhf(x);
}
#else
__device__ __forceinline__ float tanh_hw(float x) {
    x = fminf(fmaxf(x, -15.f), 15.f);
    float e = __expf(2.f * x);
    return (e - 1.f) * __builtin_amdgcn_rcpf(e + 1.f);
}
#endif

// One wave (32 lanes) handles 16 batch rows for all 512 timesteps.
// gates[16,128] = h[16,32] @ W_hh^T[32,128] + (x*W_ih + b)  via 8x v_wmma_f32_16x16x32_f16
//
// Sigmoid gates (i,f,o) use sigmoid(z) = 0.5*tanh(z/2)+0.5. The 0.5 pre-scale of
// the *argument* is linear in the weights, so it is folded into the preloaded
// B-fragments and input-projection scalars of those tiles (done once, outside
// the 512-step serial loop) -> sigmoid costs tanh + 1 fma in the loop.
__global__ __launch_bounds__(32) void lstm_wmma_kernel(
    const float* __restrict__ x,     // [B, T]
    const float* __restrict__ W_ih,  // [128, 1]
    const float* __restrict__ W_hh,  // [128, 32]
    const float* __restrict__ b_ih,  // [128]
    const float* __restrict__ b_hh,  // [128]
    const float* __restrict__ W_fc,  // [2, 32]
    const float* __restrict__ b_fc,  // [2]
    float* __restrict__ out)         // [B, 2]
{
    __shared__ _Float16 lds_h[16 * HPITCH];  // h tile, padded row-major, per-wave

    const int lid  = threadIdx.x;   // 0..31
    const int half = lid >> 4;      // lane half (WMMA layout)
    const int ln   = lid & 15;
    const int moff = half * 8;      // M offset for C/D layout
    const int b0   = blockIdx.x * 16;

    // ---- Preload W_hh^T as 8 B-fragments (f16), resident for whole loop ----
    // B[k][n] with n_global = tile*16 + ln; lane half selects K 0..15 / 16..31.
    // Tiles 4,5 are the cell gate (tanh, unscaled); all others pre-scaled 0.5.
    v16h Bf[8];
    float wv[8], bv[8];
    #pragma unroll
    for (int t = 0; t < 8; ++t) {
        const float scale = (t == 4 || t == 5) ? 1.0f : 0.5f;
        const float* wrow = W_hh + (t * 16 + ln) * HID + half * 16;
        #pragma unroll
        for (int i = 0; i < 16; ++i) Bf[t][i] = (_Float16)(scale * wrow[i]);
        const int gc = t * 16 + ln;
        wv[t] = scale * W_ih[gc];
        bv[t] = scale * (b_ih[gc] + b_hh[gc]);
    }

    // ---- State: h (A fragment, f16) = 0, c (f32, D layout) = 0 ----
    v16h a = {};
    float cst[16];
    #pragma unroll
    for (int i = 0; i < 16; ++i) cst[i] = 0.f;

    #pragma unroll 1
    for (int t = 0; t < TSTEPS; ++t) {
        // x values for this wave's 8 M-rows (per lane half)
        float xv[8];
        #pragma unroll
        for (int v = 0; v < 8; ++v)
            xv[v] = x[(b0 + moff + v) * TSTEPS + t];

        // Two column-groups; tiles {g, g+2, g+4, g+6} = i,f,g,o at same position
        #pragma unroll
        for (int g = 0; g < 2; ++g) {
            v8f D[4];
            #pragma unroll
            for (int gate = 0; gate < 4; ++gate) {
                const int tile = 2 * gate + g;
                v8f cin;
                #pragma unroll
                for (int v = 0; v < 8; ++v)
                    cin[v] = fmaf(xv[v], wv[tile], bv[tile]);
                D[gate] = __builtin_amdgcn_wmma_f32_16x16x32_f16(
                    false, a, false, Bf[tile], (short)0, cin, false, false);
            }
            // Elementwise LSTM cell update (D/C layout: m = moff+v, n = ln+16g).
            // D[0],D[1],D[3] already hold z/2 -> sigmoid = 0.5*tanh(.)+0.5.
            #pragma unroll
            for (int v = 0; v < 8; ++v) {
                float ig = fmaf(0.5f, tanh_hw(D[0][v]), 0.5f);
                float fg = fmaf(0.5f, tanh_hw(D[1][v]), 0.5f);
                float gg = tanh_hw(D[2][v]);
                float og = fmaf(0.5f, tanh_hw(D[3][v]), 0.5f);
                float c  = fmaf(fg, cst[g * 8 + v], ig * gg);
                cst[g * 8 + v] = c;
                float h = og * tanh_hw(c);
                lds_h[(moff + v) * HPITCH + (ln + 16 * g)] = (_Float16)h;
            }
        }

        // ---- Rebuild A fragment from LDS (contiguous K per lane half, same order as Bf) ----
        // lane<16: row ln, K 0..15 ; lane>=16: row ln, K 16..31
        const _Float16* rowp = &lds_h[ln * HPITCH + half * 16];
        v8h lo = *(const v8h*)(rowp);        // 16B-aligned (HPITCH*2 = 80 = 5*16)
        v8h hi = *(const v8h*)(rowp + 8);
        #pragma unroll
        for (int i = 0; i < 8; ++i) { a[i] = lo[i]; a[i + 8] = hi[i]; }
    }

    // ---- Final FC: out[b0+r][cls] = h_last[r,:] . W_fc[cls,:] + b_fc[cls] ----
    {
        const int r   = lid >> 1;
        const int cls = lid & 1;
        float s = b_fc[cls];
        #pragma unroll
        for (int k = 0; k < HID; ++k)
            s = fmaf((float)lds_h[r * HPITCH + k], W_fc[cls * HID + k], s);
        out[(b0 + r) * NCLS + cls] = s;
    }
}

extern "C" void kernel_launch(void* const* d_in, const int* in_sizes, int n_in,
                              void* d_out, int out_size, void* d_ws, size_t ws_size,
                              hipStream_t stream) {
    const float* x    = (const float*)d_in[0];
    const float* W_ih = (const float*)d_in[1];
    const float* W_hh = (const float*)d_in[2];
    const float* b_ih = (const float*)d_in[3];
    const float* b_hh = (const float*)d_in[4];
    const float* W_fc = (const float*)d_in[5];
    const float* b_fc = (const float*)d_in[6];
    float* out = (float*)d_out;

    // 64 batch tiles of 16 rows; one wave per block -> 64 independent
    // recurrence chains spread across WGPs.
    lstm_wmma_kernel<<<BATCH / 16, 32, 0, stream>>>(
        x, W_ih, W_hh, b_ih, b_hh, W_fc, b_fc, out);
}
